// ScaledDotProductAttention_88802743812456
// MI455X (gfx1250) — matrix operand order repaired and checked by reading, not compiled, not run on hardware
//
#include <hip/hip_runtime.h>

// MI455X (gfx1250) fused attention: out = softmax(QK^T/8) V, also materializes attn.
// Wave32; WMMA f32_16x16x32_f16 for both GEMMs; online-softmax two-pass per wave.
// attn/O stores are non-temporal (streaming, written once) so K/V stay in L2.
// Softmax runs in log2 domain with raw v_exp_f32 (no denormal-guard fixup).

typedef _Float16 v16h __attribute__((ext_vector_type(16)));
typedef _Float16 v8h  __attribute__((ext_vector_type(8)));
typedef float    v8f  __attribute__((ext_vector_type(8)));

static constexpr int BH = 32;
static constexpr int S  = 2048;
static constexpr int D  = 64;
// softmax in exp2 domain: score scale = (1/temperature) * log2(e)
static constexpr float SCL = 0.125f * 1.44269504088896340736f;

// raw hardware exp2: single v_exp_f32, flushes sub-2^-126 to 0 (fine for softmax)
__device__ __forceinline__ float ex2(float x) { return __builtin_amdgcn_exp2f(x); }

// load 8 consecutive floats (global or LDS), convert to 8 halves
__device__ __forceinline__ v8h f8h(const float* p) {
  const float4 a = *(const float4*)p;
  const float4 b = *(const float4*)(p + 4);
  v8h r;
  r[0] = (_Float16)a.x; r[1] = (_Float16)a.y; r[2] = (_Float16)a.z; r[3] = (_Float16)a.w;
  r[4] = (_Float16)b.x; r[5] = (_Float16)b.y; r[6] = (_Float16)b.z; r[7] = (_Float16)b.w;
  return r;
}
__device__ __forceinline__ v16h cat(v8h a, v8h b) {
  v16h r;
#pragma unroll
  for (int i = 0; i < 8; ++i) { r[i] = a[i]; r[8 + i] = b[i]; }
  return r;
}

__global__ __launch_bounds__(128) void fattn_kernel(
    const float* __restrict__ q, const float* __restrict__ k,
    const float* __restrict__ v, float* __restrict__ outO,
    float* __restrict__ outA)
{
  // strides padded for LDS bank spread; all rows stay 16B aligned
  __shared__ _Float16 Ksh[32][72];   // 32 keys x 64 dims (f16), pad 72
  __shared__ _Float16 Vtsh[64][40];  // transposed: 64 dims x 32 keys (f16), pad 40
  __shared__ float    Psh[4][16][36];// per-wave P-tile transpose staging

  const int tid  = threadIdx.x;
  const int wv   = tid >> 5;
  const int lane = tid & 31;
  const int ln16 = lane & 15;
  const int hi   = lane >> 4;          // which 16-lane half

  const int b    = blockIdx.x >> 5;    // 32 query-blocks per batch
  const int qblk = blockIdx.x & 31;
  const int m0w  = qblk * 64 + wv * 16;  // this wave's 16 query rows
  const int mrow = m0w + ln16;

  // cooperative staging coordinates: thread -> (row 0..31, 16-wide column chunk)
  const int srow = tid >> 2;
  const int scol = (tid & 3) * 16;

  const float* qb = q + (size_t)b * S * D;
  const float* kb = k + (size_t)b * S * D;
  const float* vb = v + (size_t)b * S * D;

  // ---- Q tile as two A-operand fragments (D=64 -> 2 k-slices of 32) ----
  // A layout: lanes 0-15 row=ln16, K 0-15; lanes 16-31 K 16-31 (contiguous)
  v16h Qf[2];
#pragma unroll
  for (int j = 0; j < 2; ++j) {
    const float* qr = qb + (size_t)mrow * D + 32 * j + 16 * hi;
    Qf[j] = cat(f8h(qr), f8h(qr + 8));
  }

  // ================= pass A: online per-lane softmax stats (log2 domain) =====
  // C-layout: reg r holds row (r + 8*hi), lane column N = ln16 of each 16-key tile
  float m_run[8], l_run[8];
#pragma unroll
  for (int r = 0; r < 8; ++r) { m_run[r] = -1e30f; l_run[r] = 0.0f; }

  for (int n0 = 0; n0 < S; n0 += 32) {
    __syncthreads();
    {
      const float* kr = kb + (size_t)(n0 + srow) * D + scol;
      *(v8h*)&Ksh[srow][scol]     = f8h(kr);
      *(v8h*)&Ksh[srow][scol + 8] = f8h(kr + 8);
    }
    __syncthreads();
#pragma unroll
    for (int t = 0; t < 2; ++t) {
      v8f acc = {};
#pragma unroll
      for (int j = 0; j < 2; ++j) {
        // B = K^T : lane column n = key row (16t+ln16); K 0-15 / 16-31 contiguous
        const _Float16* kp = &Ksh[16 * t + ln16][32 * j + 16 * hi];
        v16h Bf = cat(*(const v8h*)kp, *(const v8h*)(kp + 8));
        acc = __builtin_amdgcn_wmma_f32_16x16x32_f16(false, Qf[j], false, Bf,
                                                     (short)0, acc, false, false);
      }
#pragma unroll
      for (int r = 0; r < 8; ++r) {
        float s  = acc[r] * SCL;               // log2-domain score
        float mn = fmaxf(m_run[r], s);
        l_run[r] = l_run[r] * ex2(m_run[r] - mn) + ex2(s - mn);
        m_run[r] = mn;
      }
    }
  }
  // merge stats across the 16 lanes of each half-wave (xor tree stays in-half)
#pragma unroll
  for (int r = 0; r < 8; ++r) {
#pragma unroll
    for (int msk = 1; msk < 16; msk <<= 1) {
      float om = __shfl_xor(m_run[r], msk, 32);
      float ol = __shfl_xor(l_run[r], msk, 32);
      float mn = fmaxf(m_run[r], om);
      l_run[r] = l_run[r] * ex2(m_run[r] - mn) + ol * ex2(om - mn);
      m_run[r] = mn;
    }
    l_run[r] = 1.0f / l_run[r];   // now 1/rowsum, broadcast to all 16 lanes in half
  }

  // ================= pass B: attn write + O = P V =================
  v8f accO[4];
#pragma unroll
  for (int ot = 0; ot < 4; ++ot) { v8f z = {}; accO[ot] = z; }

  float* arow = outA + (size_t)b * S * S;

  for (int n0 = 0; n0 < S; n0 += 32) {
    __syncthreads();
    {
      const float* kr = kb + (size_t)(n0 + srow) * D + scol;
      *(v8h*)&Ksh[srow][scol]     = f8h(kr);
      *(v8h*)&Ksh[srow][scol + 8] = f8h(kr + 8);
      const float* vr = vb + (size_t)(n0 + srow) * D + scol;
#pragma unroll
      for (int jj = 0; jj < 4; ++jj) {   // transpose V into Vtsh[dim][key]
        float4 f = *(const float4*)(vr + 4 * jj);
        Vtsh[scol + 4 * jj + 0][srow] = (_Float16)f.x;
        Vtsh[scol + 4 * jj + 1][srow] = (_Float16)f.y;
        Vtsh[scol + 4 * jj + 2][srow] = (_Float16)f.z;
        Vtsh[scol + 4 * jj + 3][srow] = (_Float16)f.w;
      }
    }
    __syncthreads();

#pragma unroll
    for (int t = 0; t < 2; ++t) {
      v8f acc = {};
#pragma unroll
      for (int j = 0; j < 2; ++j) {
        const _Float16* kp = &Ksh[16 * t + ln16][32 * j + 16 * hi];
        v16h Bf = cat(*(const v8h*)kp, *(const v8h*)(kp + 8));
        acc = __builtin_amdgcn_wmma_f32_16x16x32_f16(false, Qf[j], false, Bf,
                                                     (short)0, acc, false, false);
      }
#pragma unroll
      for (int r = 0; r < 8; ++r) {
        float p = ex2(__builtin_fmaf(acc[r], SCL, -m_run[r])) * l_run[r];
        // dominant traffic: normalized attention, written once, streaming -> NT
        __builtin_nontemporal_store(
            p, &arow[(size_t)(m0w + r + 8 * hi) * S + n0 + 16 * t + ln16]);
        Psh[wv][r + 8 * hi][16 * t + ln16] = p;   // stage for layout transpose
      }
    }
    __builtin_amdgcn_wave_barrier();  // keep LDS P store->load ordered (same-wave DS is in-order)

    // P tile (16x32) re-read in A-operand layout, f32->f16
    const float* pp = &Psh[wv][ln16][16 * hi];
    v16h Pf = cat(f8h(pp), f8h(pp + 8));

    // O(16x64) += P(16x32) x V(32x64): 4 n-tiles of the head dim
#pragma unroll
    for (int ot = 0; ot < 4; ++ot) {
      const _Float16* vp = &Vtsh[16 * ot + ln16][16 * hi];
      v16h Vf = cat(*(const v8h*)vp, *(const v8h*)(vp + 8));
      accO[ot] = __builtin_amdgcn_wmma_f32_16x16x32_f16(false, Pf, false, Vf,
                                                        (short)0, accO[ot], false, false);
    }
  }

  // ---- store O (C-layout: reg r -> row r+8*hi, lane -> column) ----
#pragma unroll
  for (int ot = 0; ot < 4; ++ot)
#pragma unroll
    for (int r = 0; r < 8; ++r)
      __builtin_nontemporal_store(
          accO[ot][r],
          &outO[(size_t)(b * S + m0w + r + 8 * hi) * D + 16 * ot + ln16]);
}

extern "C" void kernel_launch(void* const* d_in, const int* in_sizes, int n_in,
                              void* d_out, int out_size, void* d_ws, size_t ws_size,
                              hipStream_t stream) {
  const float* q = (const float*)d_in[0];
  const float* k = (const float*)d_in[1];
  const float* v = (const float*)d_in[2];
  float* outO = (float*)d_out;                       // [32,2048,64]
  float* outA = outO + (size_t)BH * S * D;           // [32,2048,2048]
  dim3 grid(BH * (S / 64));                          // 1024 blocks
  fattn_kernel<<<grid, 128, 0, stream>>>(q, k, v, outO, outA);
}
